// SPPolicy_86234353369881
// MI455X (gfx1250) — matrix operand order repaired and checked by reading, not compiled, not run on hardware
//
#include <hip/hip_runtime.h>
#include <hip/hip_bf16.h>
#include <math.h>

typedef __bf16 bf16;
typedef __attribute__((ext_vector_type(16))) __bf16 v16bf;
typedef __attribute__((ext_vector_type(8)))  __bf16 v8bf;
typedef __attribute__((ext_vector_type(2)))  __bf16 v2bf;
typedef __attribute__((ext_vector_type(8)))  float  v8f;
typedef __attribute__((ext_vector_type(4)))  int    v4i;

#define NN    4096
#define DD    128
#define NHEAD 4
#define HDIM  32

// gfx1250 async global->LDS path (guarded; falls back to VGPR-mediated copy)
#if defined(__has_builtin)
#if __has_builtin(__builtin_amdgcn_global_load_async_to_lds_b128) && \
    __has_builtin(__builtin_amdgcn_s_wait_asynccnt)
#define HAVE_ASYNC_LDS 1
#endif
#endif
#ifndef HAVE_ASYNC_LDS
#define HAVE_ASYNC_LDS 0
#endif

#if HAVE_ASYNC_LDS
typedef __attribute__((address_space(1))) v4i as1_v4i;
typedef __attribute__((address_space(3))) v4i as3_v4i;
#endif

__device__ __forceinline__ void cp16_g2l(bf16* dst_lds, const bf16* src_g) {
#if HAVE_ASYNC_LDS
  __builtin_amdgcn_global_load_async_to_lds_b128(
      (as1_v4i*)(void*)(src_g),
      (as3_v4i*)(void*)(dst_lds), 0, 0);
#else
  *(v8bf*)dst_lds = *(const v8bf*)src_g;
#endif
}
__device__ __forceinline__ void cp_wait() {
#if HAVE_ASYNC_LDS
  __builtin_amdgcn_s_wait_asynccnt(0);
#endif
}

__device__ __forceinline__ v16bf cat8(v8bf a, v8bf b) {
  return __builtin_shufflevector(a, b, 0,1,2,3,4,5,6,7,8,9,10,11,12,13,14,15);
}
__device__ __forceinline__ v8f wmma_bf16(v16bf a, v16bf b, v8f c) {
  // D = A(16x32) * B(32x16) + C, f32 accum
  return __builtin_amdgcn_wmma_f32_16x16x32_bf16(false, a, false, b, (short)0, c, false, false);
}

// ---------------------------------------------------------------------------
// deg[i] = max(sum_j A[i,j], 1e-8)
// ---------------------------------------------------------------------------
__global__ __launch_bounds__(256) void deg_kernel(const float* __restrict__ A,
                                                  float* __restrict__ deg, int n) {
  __shared__ float red[256];
  int row = blockIdx.x, tid = threadIdx.x;
  float s = 0.f;
  for (int j = tid; j < n; j += 256) s += A[(size_t)row * n + j];
  red[tid] = s; __syncthreads();
  for (int o = 128; o > 0; o >>= 1) { if (tid < o) red[tid] += red[tid + o]; __syncthreads(); }
  if (tid == 0) deg[row] = fmaxf(red[0], 1e-8f);
}

// ---------------------------------------------------------------------------
// Anorm_bf16[i,j] = bf16(A[i,j] / deg[i])
// ---------------------------------------------------------------------------
__global__ __launch_bounds__(256) void anorm_kernel(const float* __restrict__ A,
                                                    const float* __restrict__ deg,
                                                    bf16* __restrict__ out, int n) {
  int row = blockIdx.x;
  float inv = 1.0f / deg[row];
  for (int j = threadIdx.x; j < n; j += 256)
    out[(size_t)row * n + j] = (bf16)(A[(size_t)row * n + j] * inv);
}

// ---------------------------------------------------------------------------
// GCN layer 1 (inner dim 5): msg = (A/deg) @ X ; H = relu(X Wself^T + msg Wmsg^T)
// ---------------------------------------------------------------------------
__global__ __launch_bounds__(256) void gcn1_kernel(const float* __restrict__ A,
                                                   const float* __restrict__ deg,
                                                   const float* __restrict__ X,
                                                   const float* __restrict__ wself,
                                                   const float* __restrict__ wmsg,
                                                   float* __restrict__ H, int n) {
  __shared__ float red[5][256];
  __shared__ float msgv[5];
  int row = blockIdx.x, tid = threadIdx.x;
  float invd = 1.0f / deg[row];
  float acc[5] = {0.f, 0.f, 0.f, 0.f, 0.f};
  for (int j = tid; j < n; j += 256) {
    float a = A[(size_t)row * n + j] * invd;
    #pragma unroll
    for (int c = 0; c < 5; ++c) acc[c] += a * X[j * 5 + c];
  }
  #pragma unroll
  for (int c = 0; c < 5; ++c) red[c][tid] = acc[c];
  __syncthreads();
  for (int o = 128; o > 0; o >>= 1) {
    if (tid < o) {
      #pragma unroll
      for (int c = 0; c < 5; ++c) red[c][tid] += red[c][tid + o];
    }
    __syncthreads();
  }
  if (tid < 5) msgv[tid] = red[tid][0];
  __syncthreads();
  if (tid < DD) {
    float h = 0.f;
    #pragma unroll
    for (int c = 0; c < 5; ++c)
      h += X[row * 5 + c] * wself[tid * 5 + c] + msgv[c] * wmsg[tid * 5 + c];
    H[(size_t)row * DD + tid] = fmaxf(h, 0.f);
  }
}

// ---------------------------------------------------------------------------
// BatchNorm (column stats) -> f32 + row-major bf16 + optional transposed bf16
// ---------------------------------------------------------------------------
__global__ __launch_bounds__(256) void bn_kernel(const float* __restrict__ H,
                                                 const float* __restrict__ g,
                                                 const float* __restrict__ b,
                                                 float* __restrict__ out,
                                                 bf16* __restrict__ outbf,
                                                 bf16* __restrict__ outbfT,
                                                 int n, int d) {
  __shared__ float rs[256], rs2[256];
  int col = blockIdx.x, tid = threadIdx.x;
  float s = 0.f, s2 = 0.f;
  for (int i = tid; i < n; i += 256) {
    float v = H[(size_t)i * d + col];
    s += v; s2 += v * v;
  }
  rs[tid] = s; rs2[tid] = s2; __syncthreads();
  for (int o = 128; o > 0; o >>= 1) {
    if (tid < o) { rs[tid] += rs[tid + o]; rs2[tid] += rs2[tid + o]; }
    __syncthreads();
  }
  float mean = rs[0] / n;
  float var  = rs2[0] / n - mean * mean;
  float sc = rsqrtf(var + 1e-5f) * g[col];
  float sh = b[col] - mean * sc;
  for (int i = tid; i < n; i += 256) {
    float v = H[(size_t)i * d + col] * sc + sh;
    out[(size_t)i * d + col] = v;
    outbf[(size_t)i * d + col] = (bf16)v;
    if (outbfT) outbfT[(size_t)col * n + i] = (bf16)v;
  }
}

// ---------------------------------------------------------------------------
// LayerNorm over last dim (d=128): out = LN(x + resid)*g + b, + bf16 copy
// ---------------------------------------------------------------------------
__global__ __launch_bounds__(128) void ln_kernel(const float* __restrict__ x,
                                                 const float* __restrict__ resid,
                                                 const float* __restrict__ g,
                                                 const float* __restrict__ b,
                                                 float* __restrict__ out,
                                                 bf16* __restrict__ outbf, int d) {
  __shared__ float red[128];
  int row = blockIdx.x, tid = threadIdx.x;
  float v = x[(size_t)row * d + tid];
  if (resid) v += resid[(size_t)row * d + tid];
  red[tid] = v; __syncthreads();
  for (int o = 64; o > 0; o >>= 1) { if (tid < o) red[tid] += red[tid + o]; __syncthreads(); }
  float mean = red[0] / d;
  __syncthreads();
  float c = v - mean;
  red[tid] = c * c; __syncthreads();
  for (int o = 64; o > 0; o >>= 1) { if (tid < o) red[tid] += red[tid + o]; __syncthreads(); }
  float var = red[0] / d;
  float y = c * rsqrtf(var + 1e-5f) * g[tid] + b[tid];
  out[(size_t)row * d + tid] = y;
  if (outbf) outbf[(size_t)row * d + tid] = (bf16)y;
}

// ---------------------------------------------------------------------------
// plain cast: W (f32, [outd][ind]) -> bf16 same layout == BT[N][K] operand.
// ---------------------------------------------------------------------------
__global__ __launch_bounds__(256) void wcast_kernel(const float* __restrict__ W,
                                                    bf16* __restrict__ out, int tot) {
  int i = blockIdx.x * 256 + threadIdx.x;
  if (i < tot) out[i] = (bf16)W[i];
}

// ---------------------------------------------------------------------------
// WMMA GEMM: C[M,N] = A[M,K](bf16,row-major) x BT[N,K](bf16,row-major)^T
// Block = 8 waves, 64x128 C tile, K-step 32: each wave reuses one B fragment
// across 4 WMMAs. LDS staged with gfx1250 async global->LDS when available.
// ---------------------------------------------------------------------------
__global__ __launch_bounds__(256) void wmma_gemm_kernel(
    const bf16* __restrict__ A, int lda,
    const bf16* __restrict__ BT, int ldbt,
    const float* __restrict__ bias,
    const float* __restrict__ addsrc,
    float* __restrict__ C, bf16* __restrict__ Cbf, int ldc,
    int M, int N, int K, int relu) {
  __shared__ bf16 As[64 * 32];    // A tile   [64 rows][32 k]
  __shared__ bf16 Bs[128 * 32];   // BT tile  [128 n ][32 k]
  int tid = threadIdx.x;
  int wave = tid >> 5;
  int lane = tid & 31;
  int r15 = lane & 15, hi = lane >> 4;
  int rowbase = blockIdx.x * 64;
  int colblk  = blockIdx.y * 128;

  v8f acc[4] = {{}, {}, {}, {}};
  for (int k0 = 0; k0 < K; k0 += 32) {
    // stage A: 2048 elems = 256 threads x 8 (16B)
    {
      int e = tid * 8;
      int r = e >> 5, c = e & 31;
      cp16_g2l(&As[e], &A[(size_t)(rowbase + r) * lda + k0 + c]);
    }
    // stage BT: 4096 elems = 256 threads x 2 x 8 (16B)
    #pragma unroll
    for (int ch = 0; ch < 2; ++ch) {
      int c2 = tid * 2 + ch;           // 0..511
      int nr = c2 >> 2;                // 0..127
      int kc = (c2 & 3) * 8;           // 0,8,16,24
      cp16_g2l(&Bs[nr * 32 + kc], &BT[(size_t)(colblk + nr) * ldbt + k0 + kc]);
    }
    if (k0 + 32 < K) {   // warm next tiles toward L2
      __builtin_prefetch(&A[(size_t)(rowbase + (tid >> 2)) * lda + k0 + 32 + (tid & 3) * 8], 0, 1);
      __builtin_prefetch(&BT[(size_t)(colblk + (tid >> 1)) * ldbt + k0 + 32 + (tid & 1) * 16], 0, 1);
    }
    cp_wait();
    __syncthreads();
    // B fragment (shared across this wave's 4 row-tiles)
    int ncol = wave * 16 + r15;
    v16bf bfrag = cat8(*(const v8bf*)&Bs[ncol * 32 + hi * 8],
                       *(const v8bf*)&Bs[ncol * 32 + hi * 8 + 16]);
    #pragma unroll
    for (int rt = 0; rt < 4; ++rt) {
      int ar = rt * 16 + r15;
      v16bf afrag = cat8(*(const v8bf*)&As[ar * 32 + hi * 8],
                         *(const v8bf*)&As[ar * 32 + hi * 8 + 16]);
      acc[rt] = wmma_bf16(afrag, bfrag, acc[rt]);
    }
    __syncthreads();
  }
  // epilogue
  int colg = colblk + wave * 16 + r15;
  #pragma unroll
  for (int rt = 0; rt < 4; ++rt) {
    #pragma unroll
    for (int v = 0; v < 8; ++v) {
      int rowg = rowbase + rt * 16 + v + hi * 8;
      float x = acc[rt][v];
      if (bias)   x += bias[colg];
      if (addsrc) x += addsrc[(size_t)rowg * ldc + colg];
      if (relu)   x = fmaxf(x, 0.f);
      if (C)   C[(size_t)rowg * ldc + colg] = x;
      if (Cbf) Cbf[(size_t)rowg * ldc + colg] = (bf16)x;
    }
  }
}

// ---------------------------------------------------------------------------
// Flash-style attention. qkv: [N][384] bf16 (q|k|v each 128 wide, 4 heads x 32).
// One wave per (16-query tile, head). Online softmax; P transposed via LDS.
// ---------------------------------------------------------------------------
__global__ __launch_bounds__(32) void attn_kernel(const bf16* __restrict__ qkv,
                                                  bf16* __restrict__ obf, int n) {
  const int ld = 3 * DD;
  int qt = blockIdx.x, h = blockIdx.y;
  int lane = threadIdx.x;
  int r15 = lane & 15, hi = lane >> 4;
  const float scale = 0.1767766952966369f;  // 1/sqrt(32)

  const bf16* Q  = qkv + h * HDIM;
  const bf16* Kp = qkv + DD + h * HDIM;
  const bf16* Vp = qkv + 2 * DD + h * HDIM;

  int qrow = qt * 16 + r15;
  v16bf qf = cat8(*(const v8bf*)&Q[(size_t)qrow * ld + hi * 8],
                  *(const v8bf*)&Q[(size_t)qrow * ld + hi * 8 + 16]);

  float mrow[8], lrow[8];
  v8f o0 = {}, o1 = {};
  #pragma unroll
  for (int i = 0; i < 8; ++i) { mrow[i] = -1e30f; lrow[i] = 0.f; }

  __shared__ bf16 Pt[16 * 32];

  for (int kc = 0; kc < n; kc += 32) {
    v8f s[2];
    #pragma unroll
    for (int j = 0; j < 2; ++j) {
      int key = kc + j * 16 + r15;
      v16bf kf = cat8(*(const v8bf*)&Kp[(size_t)key * ld + hi * 8],
                      *(const v8bf*)&Kp[(size_t)key * ld + hi * 8 + 16]);
      v8f z = {};
      s[j] = wmma_bf16(qf, kf, z);
    }
    #pragma unroll
    for (int v = 0; v < 8; ++v) {
      float a0 = s[0][v] * scale, a1 = s[1][v] * scale;
      float tmx = fmaxf(a0, a1);
      tmx = fmaxf(tmx, __shfl_xor(tmx, 1, 32));
      tmx = fmaxf(tmx, __shfl_xor(tmx, 2, 32));
      tmx = fmaxf(tmx, __shfl_xor(tmx, 4, 32));
      tmx = fmaxf(tmx, __shfl_xor(tmx, 8, 32));
      float mnew = fmaxf(mrow[v], tmx);
      float corr = expf(mrow[v] - mnew);
      float p0 = expf(a0 - mnew), p1 = expf(a1 - mnew);
      float rsum = p0 + p1;
      rsum += __shfl_xor(rsum, 1, 32);
      rsum += __shfl_xor(rsum, 2, 32);
      rsum += __shfl_xor(rsum, 4, 32);
      rsum += __shfl_xor(rsum, 8, 32);
      lrow[v] = lrow[v] * corr + rsum;
      mrow[v] = mnew;
      o0[v] *= corr; o1[v] *= corr;
      s[0][v] = p0; s[1][v] = p1;
    }
    #pragma unroll
    for (int v = 0; v < 8; ++v) {
      int prow = v + hi * 8;
      Pt[prow * 32 + r15]      = (bf16)s[0][v];
      Pt[prow * 32 + 16 + r15] = (bf16)s[1][v];
    }
    __syncthreads();
    v16bf pf = cat8(*(const v8bf*)&Pt[r15 * 32 + hi * 16],
                    *(const v8bf*)&Pt[r15 * 32 + hi * 16 + 8]);
    __syncthreads();
    #pragma unroll
    for (int jn = 0; jn < 2; ++jn) {
      v16bf vf;
      #pragma unroll
      for (int i = 0; i < 16; ++i) {
        int key = kc + hi * 16 + i;
        vf[i] = Vp[(size_t)key * ld + jn * 16 + r15];
      }
      if (jn == 0) o0 = wmma_bf16(pf, vf, o0);
      else         o1 = wmma_bf16(pf, vf, o1);
    }
  }
  #pragma unroll
  for (int v = 0; v < 8; ++v) {
    int row = qt * 16 + v + hi * 8;
    float inv = 1.0f / lrow[v];
    obf[(size_t)row * DD + h * HDIM + r15]      = (bf16)(o0[v] * inv);
    obf[(size_t)row * DD + h * HDIM + 16 + r15] = (bf16)(o1[v] * inv);
  }
}

// ---------------------------------------------------------------------------
// threefry2x32-20 + helpers for the pointer decoder's Gumbel sampling
// ---------------------------------------------------------------------------
__device__ __forceinline__ unsigned rotl32(unsigned x, int r) {
  return (x << r) | (x >> (32 - r));
}
__device__ __forceinline__ void threefry2x32(unsigned k0, unsigned k1,
                                             unsigned x0, unsigned x1,
                                             unsigned& o0, unsigned& o1) {
  const unsigned rot[8] = {13, 15, 26, 6, 17, 29, 16, 24};
  unsigned ks[3] = {k0, k1, k0 ^ k1 ^ 0x1BD11BDAu};
  unsigned X0 = x0 + ks[0], X1 = x1 + ks[1];
  #pragma unroll
  for (unsigned b = 1; b <= 5; ++b) {
    #pragma unroll
    for (int j = 0; j < 4; ++j) {
      X0 += X1;
      X1 = rotl32(X1, rot[((b - 1) & 1) * 4 + j]);
      X1 ^= X0;
    }
    X0 += ks[b % 3];
    X1 += ks[(b + 1) % 3] + b;
  }
  o0 = X0; o1 = X1;
}
__device__ __forceinline__ float u01(unsigned bits) {
  float r = __uint_as_float(0x3f800000u | (bits >> 9)) - 1.0f;
  return fmaxf(r, 1e-12f);
}

// ---------------------------------------------------------------------------
// Pointer decoder: whole 64-step loop in one block (256 threads).
// ---------------------------------------------------------------------------
__global__ __launch_bounds__(256) void decoder_kernel(
    const float* __restrict__ emb,
    const float* __restrict__ Kd,
    const float* __restrict__ adj,
    const float* __restrict__ wq,
    const float* __restrict__ wih, const float* __restrict__ bih,
    const float* __restrict__ whh, const float* __restrict__ bhh,
    const int* __restrict__ sp, const int* __restrict__ tp,
    const int* __restrict__ msp,
    float* __restrict__ out, int n) {
  const float LOG_EPS = -20.72326583694641f;
  __shared__ float hid[DD], qv[DD], gi[3 * DD], gh[3 * DD];
  __shared__ float sc[NN];
  __shared__ float red[256];
  __shared__ int   redi[256];
  __shared__ int   s_cur, s_done, s_nxt, s_dead;
  __shared__ float s_lp;
  int tid = threadIdx.x;
  int s = sp[0], t = tp[0];
  int msteps = msp[0]; if (msteps > 64) msteps = 64;
  if (tid == 0) { s_cur = s; s_done = 0; s_lp = 0.f; }
  if (tid < DD) hid[tid] = emb[(size_t)s * DD + tid];
  for (int i = tid; i < 65; i += 256) out[i] = (i == 0) ? (float)s : -1.0f;
  __syncthreads();

  for (int step = 0; step < 64; ++step) {
    if (step >= msteps) break;
    int cur = s_cur;
    int done0 = s_done;
    if (tid < DD) {
      float a = 0.f;
      for (int k = 0; k < DD; ++k) a += wq[tid * DD + k] * hid[k];
      qv[tid] = a;
    }
    __syncthreads();
    unsigned fk0, fk1;
    threefry2x32(0u, 1u, 0u, (unsigned)step, fk0, fk1);
    const float qscale = 0.08838834764831845f;  // 1/sqrt(128)
    float best = -3e38f; int besti = 0; int anyn = 0;
    for (int i = tid; i < n; i += 256) {
      const float* kr = Kd + (size_t)i * DD;
      float d = 0.f;
      for (int k = 0; k < DD; ++k) d += kr[k] * qv[k];
      d *= qscale;
      bool nbr = (adj[(size_t)cur * n + i] > 0.f) && (i != cur);
      float logit = nbr ? d : -1e9f;
      sc[i] = logit;
      anyn |= (int)nbr;
      unsigned r0, r1;
      threefry2x32(fk0, fk1, 0u, (unsigned)i, r0, r1);
      float gum = -logf(-logf(u01(r0)));
      float val = logit + gum;
      if (val > best) { best = val; besti = i; }
    }
    red[tid] = best; redi[tid] = besti; __syncthreads();
    for (int o = 128; o > 0; o >>= 1) {
      if (tid < o && red[tid + o] > red[tid]) { red[tid] = red[tid + o]; redi[tid] = redi[tid + o]; }
      __syncthreads();
    }
    if (tid == 0) s_nxt = redi[0];
    __syncthreads();
    redi[tid] = anyn; __syncthreads();
    for (int o = 128; o > 0; o >>= 1) {
      if (tid < o) redi[tid] |= redi[tid + o];
      __syncthreads();
    }
    if (tid == 0) s_dead = (redi[0] == 0);
    __syncthreads();
    int dead0 = s_dead;
    int nxt = s_nxt;
    int active = !(done0 || dead0);
    float mx = -3e38f;
    for (int i = tid; i < n; i += 256) mx = fmaxf(mx, sc[i]);
    red[tid] = mx; __syncthreads();
    for (int o = 128; o > 0; o >>= 1) { if (tid < o) red[tid] = fmaxf(red[tid], red[tid + o]); __syncthreads(); }
    float M = red[0]; __syncthreads();
    float se = 0.f;
    for (int i = tid; i < n; i += 256) se += expf(sc[i] - M);
    red[tid] = se; __syncthreads();
    for (int o = 128; o > 0; o >>= 1) { if (tid < o) red[tid] += red[tid + o]; __syncthreads(); }
    float lpn = sc[nxt] - M - logf(red[0]);
    __syncthreads();
    const float* xv = emb + (size_t)nxt * DD;
    for (int o = tid; o < 3 * DD; o += 256) {
      float a = bih[o], b = bhh[o];
      for (int k = 0; k < DD; ++k) {
        a += wih[o * DD + k] * xv[k];
        b += whh[o * DD + k] * hid[k];
      }
      gi[o] = a; gh[o] = b;
    }
    __syncthreads();
    if (tid < DD) {
      float r = 1.f / (1.f + expf(-(gi[tid] + gh[tid])));
      float z = 1.f / (1.f + expf(-(gi[DD + tid] + gh[DD + tid])));
      float nn_ = tanhf(gi[2 * DD + tid] + r * gh[2 * DD + tid]);
      float hnew = (1.f - z) * nn_ + z * hid[tid];
      if (active) hid[tid] = hnew;
    }
    if (tid == 0) {
      int newcur = active ? nxt : cur;
      if (active) {
        s_cur = nxt;
        out[1 + step] = (float)nxt;
        s_lp += fmaxf(lpn, LOG_EPS);
      }
      if (dead0 || newcur == t) s_done = 1;
    }
    __syncthreads();
  }
  if (tid == 0) out[65] = s_lp;
}

// ---------------------------------------------------------------------------
// host side
// ---------------------------------------------------------------------------
static void launch_gemm(const bf16* A, int lda, const bf16* BT, int ldbt,
                        const float* bias, const float* addsrc,
                        float* C, bf16* Cbf, int ldc,
                        int M, int N, int K, int relu, hipStream_t stream) {
  dim3 g(M / 64, N / 128);
  wmma_gemm_kernel<<<g, 256, 0, stream>>>(A, lda, BT, ldbt, bias, addsrc, C, Cbf, ldc, M, N, K, relu);
}
static void launch_wcast(const float* W, bf16* out, int tot, hipStream_t stream) {
  wcast_kernel<<<(tot + 255) / 256, 256, 0, stream>>>(W, out, tot);
}

extern "C" void kernel_launch(void* const* d_in, const int* in_sizes, int n_in,
                              void* d_out, int out_size, void* d_ws, size_t ws_size,
                              hipStream_t stream) {
  (void)in_sizes; (void)n_in; (void)out_size; (void)ws_size;
  const int n = NN;
  const float* node_feats = (const float*)d_in[0];
  const float* adj        = (const float*)d_in[1];
  const float* g0_wself = (const float*)d_in[2];
  const float* g0_wmsg  = (const float*)d_in[3];
  const float* g0_bng   = (const float*)d_in[4];
  const float* g0_bnb   = (const float*)d_in[5];
  const float* g1_wself = (const float*)d_in[6];
  const float* g1_wmsg  = (const float*)d_in[7];
  const float* g1_bng   = (const float*)d_in[8];
  const float* g1_bnb   = (const float*)d_in[9];
  const float* dec_wq  = (const float*)d_in[34];
  const float* dec_wk  = (const float*)d_in[35];
  const float* dec_wih = (const float*)d_in[36];
  const float* dec_bih = (const float*)d_in[37];
  const float* dec_whh = (const float*)d_in[38];
  const float* dec_bhh = (const float*)d_in[39];
  const int* s_in  = (const int*)d_in[40];
  const int* t_in  = (const int*)d_in[41];
  const int* ms_in = (const int*)d_in[42];

  char* ws = (char*)d_ws;
  size_t off = 0;
  auto alloc = [&](size_t bytes) -> void* {
    void* p = ws + off;
    off = (off + bytes + 255) & ~(size_t)255;
    return p;
  };
  float* deg    = (float*)alloc((size_t)n * 4);
  bf16*  anorm  = (bf16*) alloc((size_t)n * n * 2);
  float* h1     = (float*)alloc((size_t)n * DD * 4);
  float* h1n    = (float*)alloc((size_t)n * DD * 4);
  bf16*  h1nbf  = (bf16*) alloc((size_t)n * DD * 2);
  bf16*  h1nT   = (bf16*) alloc((size_t)n * DD * 2);   // [128][4096] transposed
  float* msg2   = (float*)alloc((size_t)n * DD * 4);
  bf16*  msg2bf = (bf16*) alloc((size_t)n * DD * 2);
  float* tmp1   = (float*)alloc((size_t)n * DD * 4);
  float* h2     = (float*)alloc((size_t)n * DD * 4);
  float* x      = (float*)alloc((size_t)n * DD * 4);
  bf16*  xbf    = (bf16*) alloc((size_t)n * DD * 2);
  float* qkvf   = (float*)alloc((size_t)n * 3 * DD * 4);
  bf16*  qkvbf  = (bf16*) alloc((size_t)n * 3 * DD * 2);
  bf16*  obf    = (bf16*) alloc((size_t)n * DD * 2);
  float* tmpo   = (float*)alloc((size_t)n * DD * 4);
  float* x1     = (float*)alloc((size_t)n * DD * 4);
  bf16*  x1bf   = (bf16*) alloc((size_t)n * DD * 2);
  float* ff1    = (float*)alloc((size_t)n * 2 * DD * 4);
  bf16*  ff1bf  = (bf16*) alloc((size_t)n * 2 * DD * 2);
  float* kdec   = (float*)alloc((size_t)n * DD * 4);
  // bf16 weight copies ([out][in] layout == BT[N][K] operand)
  bf16* wself2bf = (bf16*)alloc((size_t)DD * DD * 2);
  bf16* wmsg2bf  = (bf16*)alloc((size_t)DD * DD * 2);
  bf16* wkbf     = (bf16*)alloc((size_t)DD * DD * 2);
  bf16 *inwbf[2], *outwbf[2], *l1wbf[2], *l2wbf[2];
  for (int L = 0; L < 2; ++L) {
    inwbf[L]  = (bf16*)alloc((size_t)3 * DD * DD * 2);
    outwbf[L] = (bf16*)alloc((size_t)DD * DD * 2);
    l1wbf[L]  = (bf16*)alloc((size_t)2 * DD * DD * 2);
    l2wbf[L]  = (bf16*)alloc((size_t)DD * 2 * DD * 2);
  }

  // ---- weight casts (tiny, no transpose needed) ----
  launch_wcast(g1_wself, wself2bf, DD * DD, stream);
  launch_wcast(g1_wmsg,  wmsg2bf,  DD * DD, stream);
  launch_wcast(dec_wk,   wkbf,     DD * DD, stream);
  for (int L = 0; L < 2; ++L) {
    const int base = 10 + 12 * L;
    launch_wcast((const float*)d_in[base + 0], inwbf[L],  3 * DD * DD, stream);
    launch_wcast((const float*)d_in[base + 2], outwbf[L], DD * DD, stream);
    launch_wcast((const float*)d_in[base + 6], l1wbf[L],  2 * DD * DD, stream);
    launch_wcast((const float*)d_in[base + 8], l2wbf[L],  DD * 2 * DD, stream);
  }

  // ---- GCN ----
  deg_kernel<<<n, 256, 0, stream>>>(adj, deg, n);
  anorm_kernel<<<n, 256, 0, stream>>>(adj, deg, anorm, n);
  gcn1_kernel<<<n, 256, 0, stream>>>(adj, deg, node_feats, g0_wself, g0_wmsg, h1, n);
  bn_kernel<<<DD, 256, 0, stream>>>(h1, g0_bng, g0_bnb, h1n, h1nbf, h1nT, n, DD);
  // msg2 = Anorm @ H1n   (M=4096, N=128, K=4096; BT = h1nT [128][4096])
  launch_gemm(anorm, n, h1nT, n, nullptr, nullptr, msg2, msg2bf, DD, n, DD, n, 0, stream);
  // h2 = relu(H1n Wself^T + msg2 Wmsg^T)
  launch_gemm(h1nbf, DD, wself2bf, DD, nullptr, nullptr, tmp1, nullptr, DD, n, DD, DD, 0, stream);
  launch_gemm(msg2bf, DD, wmsg2bf, DD, nullptr, tmp1, h2, nullptr, DD, n, DD, DD, 1, stream);
  bn_kernel<<<DD, 256, 0, stream>>>(h2, g1_bng, g1_bnb, x, xbf, nullptr, n, DD);

  // ---- transformer layers ----
  for (int L = 0; L < 2; ++L) {
    const int base = 10 + 12 * L;
    const float* in_b  = (const float*)d_in[base + 1];
    const float* out_b = (const float*)d_in[base + 3];
    const float* ln1g  = (const float*)d_in[base + 4];
    const float* ln1b  = (const float*)d_in[base + 5];
    const float* l1_b  = (const float*)d_in[base + 7];
    const float* l2_b  = (const float*)d_in[base + 9];
    const float* ln2g  = (const float*)d_in[base + 10];
    const float* ln2b  = (const float*)d_in[base + 11];
    launch_gemm(xbf, DD, inwbf[L], DD, in_b, nullptr, qkvf, qkvbf, 3 * DD, n, 3 * DD, DD, 0, stream);
    attn_kernel<<<dim3(n / 16, NHEAD), 32, 0, stream>>>(qkvbf, obf, n);
    launch_gemm(obf, DD, outwbf[L], DD, out_b, nullptr, tmpo, nullptr, DD, n, DD, DD, 0, stream);
    ln_kernel<<<n, 128, 0, stream>>>(x, tmpo, ln1g, ln1b, x1, x1bf, DD);
    launch_gemm(x1bf, DD, l1wbf[L], DD, l1_b, nullptr, ff1, ff1bf, 2 * DD, n, 2 * DD, DD, 1, stream);
    launch_gemm(ff1bf, 2 * DD, l2wbf[L], 2 * DD, l2_b, nullptr, tmpo, nullptr, DD, n, DD, 2 * DD, 0, stream);
    ln_kernel<<<n, 128, 0, stream>>>(x1, tmpo, ln2g, ln2b, x, xbf, DD);
  }

  // ---- decoder ----
  launch_gemm(xbf, DD, wkbf, DD, nullptr, nullptr, kdec, nullptr, DD, n, DD, DD, 0, stream);
  decoder_kernel<<<1, 256, 0, stream>>>(x, kdec, adj, dec_wq, dec_wih, dec_bih,
                                        dec_whh, dec_bhh, s_in, t_in, ms_in,
                                        (float*)d_out, n);
}
